// AdjacencyBiasedBlock_6098853560441
// MI455X (gfx1250) — matrix operand order repaired and checked by reading, not compiled, not run on hardware
//
#include <hip/hip_runtime.h>

#define BB 4
#define NNN 1024
#define DD 768
#define HH 12
#define DHH 64
#define HID 3072
#define TT (BB * NNN)

typedef __attribute__((ext_vector_type(16))) __bf16 v16bf;
typedef __attribute__((ext_vector_type(8)))  float  v8f;

__device__ __forceinline__ float gelu_exact(float x) {
  return 0.5f * x * (1.0f + erff(x * 0.70710678118654752f));
}

// DPP16 lane swizzle, full-VALU (no LDS): ctrl 0xB1=xor1, 0x4E=xor2,
// 0x141=row_half_mirror (~xor4 once 4-groups uniform), 0x140=row_mirror (~xor8).
template <int CTRL>
__device__ __forceinline__ float dpp_movf(float x) {
  return __int_as_float(
      __builtin_amdgcn_update_dpp(0, __float_as_int(x), CTRL, 0xf, 0xf, true));
}
__device__ __forceinline__ float row16_max(float v) {
  v = fmaxf(v, dpp_movf<0xB1>(v));
  v = fmaxf(v, dpp_movf<0x4E>(v));
  v = fmaxf(v, dpp_movf<0x141>(v));
  v = fmaxf(v, dpp_movf<0x140>(v));
  return v;
}
__device__ __forceinline__ float row16_sum(float v) {
  v += dpp_movf<0xB1>(v);
  v += dpp_movf<0x4E>(v);
  v += dpp_movf<0x141>(v);
  v += dpp_movf<0x140>(v);
  return v;
}

// ---------------- fp32 -> bf16 elementwise ----------------
__global__ void cvt_bf16_kernel(const float* __restrict__ in, __bf16* __restrict__ out, int n) {
  int i = blockIdx.x * 256 + threadIdx.x;
  if (i < n) out[i] = (__bf16)in[i];
}

// ---------------- LayerNorm: fp32 row -> bf16 row ----------------
__global__ void ln_kernel(const float* __restrict__ x, const float* __restrict__ g,
                          const float* __restrict__ b, __bf16* __restrict__ out) {
  int row = blockIdx.x;
  int tid = threadIdx.x;
  const float* xr = x + (size_t)row * DD;
  float v0 = xr[tid], v1 = xr[tid + 256], v2 = xr[tid + 512];
  __shared__ float red[256];
  red[tid] = v0 + v1 + v2;
  __syncthreads();
  for (int off = 128; off > 0; off >>= 1) {
    if (tid < off) red[tid] += red[tid + off];
    __syncthreads();
  }
  float mean = red[0] * (1.0f / DD);
  __syncthreads();
  float d0 = v0 - mean, d1 = v1 - mean, d2 = v2 - mean;
  red[tid] = d0 * d0 + d1 * d1 + d2 * d2;
  __syncthreads();
  for (int off = 128; off > 0; off >>= 1) {
    if (tid < off) red[tid] += red[tid + off];
    __syncthreads();
  }
  float rstd = rsqrtf(red[0] * (1.0f / DD) + 1e-6f);
  __bf16* orow = out + (size_t)row * DD;
  orow[tid]       = (__bf16)(d0 * rstd * g[tid]       + b[tid]);
  orow[tid + 256] = (__bf16)(d1 * rstd * g[tid + 256] + b[tid + 256]);
  orow[tid + 512] = (__bf16)(d2 * rstd * g[tid + 512] + b[tid + 512]);
}

// ---------------- WMMA GEMM: C = A(bf16,MxK) @ B(bf16,KxN) + bias [+epilogue] -----
// Block tile 128x64, BK=32. 8 waves = 4(m) x 2(n); each wave owns 32x32 (4 accs, 4 WMMAs/step).
// EPI: 0 = bias -> bf16 out; 1 = bias + exact GELU -> bf16 out; 2 = bias + residual -> f32 out
template <int EPI>
__global__ __launch_bounds__(256) void gemm_wmma_kernel(
    const __bf16* __restrict__ A, const __bf16* __restrict__ Bw,
    const float* __restrict__ bias, const float* __restrict__ res,
    void* __restrict__ out, int M, int Nn, int K) {
  __shared__ __align__(32) __bf16 sA[128 * 32];  // [m][k]
  __shared__ __align__(32) __bf16 sBt[64 * 32];  // [n][k] (transposed)
  int tid  = threadIdx.x;
  int wave = tid >> 5, lane = tid & 31, half = lane >> 4, l16 = lane & 15;
  int mw = wave & 3, nw = wave >> 2;  // 4 x 2 wave grid
  int m0 = blockIdx.y * 128, n0 = blockIdx.x * 64;

  v8f acc00 = {}, acc01 = {}, acc10 = {}, acc11 = {};

  int arow = tid >> 2, acol = (tid & 3) << 3;  // 128x32 A tile: rows arow, arow+64
  int brow = tid >> 3, bcol = (tid & 7) << 3;  // 32x64  B tile

  for (int k0 = 0; k0 < K; k0 += 32) {
    const __bf16* ap0 = A + (size_t)(m0 + arow) * K + k0 + acol;
    const __bf16* ap1 = A + (size_t)(m0 + arow + 64) * K + k0 + acol;
    const __bf16* bp  = Bw + (size_t)(k0 + brow) * Nn + n0 + bcol;
    uint4 av0 = *reinterpret_cast<const uint4*>(ap0);
    uint4 av1 = *reinterpret_cast<const uint4*>(ap1);
    uint4 bv  = *reinterpret_cast<const uint4*>(bp);
    if (k0 + 32 < K) {  // prefetch next K slab (global_prefetch_b8)
      __builtin_prefetch(ap0 + 32, 0, 1);
      __builtin_prefetch(ap1 + 32, 0, 1);
      __builtin_prefetch(bp + (size_t)32 * Nn, 0, 1);
    }
    *reinterpret_cast<uint4*>(&sA[arow * 32 + acol]) = av0;
    *reinterpret_cast<uint4*>(&sA[(arow + 64) * 32 + acol]) = av1;
    union { uint4 q; __bf16 e[8]; } bu; bu.q = bv;
#pragma unroll
    for (int j = 0; j < 8; ++j) sBt[(bcol + j) * 32 + brow] = bu.e[j];
    __syncthreads();

    v16bf af0, af1;
    {
      const __bf16* ap = &sA[(mw * 32 + l16) * 32 + half * 16];
      reinterpret_cast<uint4*>(&af0)[0] = *reinterpret_cast<const uint4*>(ap);
      reinterpret_cast<uint4*>(&af0)[1] = *reinterpret_cast<const uint4*>(ap + 8);
      const __bf16* aq = &sA[(mw * 32 + 16 + l16) * 32 + half * 16];
      reinterpret_cast<uint4*>(&af1)[0] = *reinterpret_cast<const uint4*>(aq);
      reinterpret_cast<uint4*>(&af1)[1] = *reinterpret_cast<const uint4*>(aq + 8);
    }
    v16bf bf0 = *reinterpret_cast<const v16bf*>(&sBt[(nw * 32 + l16) * 32 + half * 16]);
    v16bf bf1 = *reinterpret_cast<const v16bf*>(&sBt[(nw * 32 + 16 + l16) * 32 + half * 16]);
    acc00 = __builtin_amdgcn_wmma_f32_16x16x32_bf16(false, af0, false, bf0, (short)0, acc00, false, false);
    acc01 = __builtin_amdgcn_wmma_f32_16x16x32_bf16(false, af0, false, bf1, (short)0, acc01, false, false);
    acc10 = __builtin_amdgcn_wmma_f32_16x16x32_bf16(false, af1, false, bf0, (short)0, acc10, false, false);
    acc11 = __builtin_amdgcn_wmma_f32_16x16x32_bf16(false, af1, false, bf1, (short)0, acc11, false, false);
    __syncthreads();
  }

#pragma unroll
  for (int mi = 0; mi < 2; ++mi) {
#pragma unroll
    for (int ni = 0; ni < 2; ++ni) {
      v8f acc = (mi == 0) ? (ni == 0 ? acc00 : acc01) : (ni == 0 ? acc10 : acc11);
      int col = n0 + nw * 32 + ni * 16 + l16;
      float bv = bias[col];
#pragma unroll
      for (int r = 0; r < 8; ++r) {
        int row = m0 + mw * 32 + mi * 16 + r + 8 * half;
        float v = acc[r] + bv;
        if (EPI == 0) {
          ((__bf16*)out)[(size_t)row * Nn + col] = (__bf16)v;
        } else if (EPI == 1) {
          ((__bf16*)out)[(size_t)row * Nn + col] = (__bf16)gelu_exact(v);
        } else {
          ((float*)out)[(size_t)row * Nn + col] = v + res[(size_t)row * Nn + col];
        }
      }
    }
  }
}

// ---------------- Flash attention with additive adjacency bias ----------------
// qkv layout: [token 4096][3*768], q cols [0,768), k [768,1536), v [1536,2304)
// block = (b, h, 128-query slab); 8 waves x 16 queries; key tiles of 32 shared via LDS.
// Softmax runs in base-2 domain: single v_exp_f32 per exponential.
// __launch_bounds__(256, 1): one workgroup per SIMD-group -> full VGPR budget, no spills.
__global__ __launch_bounds__(256, 1) void attn_kernel(
    const __bf16* __restrict__ qkv, const float* __restrict__ adj,
    __bf16* __restrict__ attn_out) {
  __shared__ __align__(32) __bf16 sK[32 * 64];   // [key][dh]
  __shared__ __align__(32) __bf16 sVt[64 * 32];  // [dh][key]
  __shared__ __align__(32) __bf16 sP[8 * 16 * 32];

  int tid  = threadIdx.x;
  int wave = tid >> 5, lane = tid & 31, half = lane >> 4, l16 = lane & 15;
  int blk = blockIdx.x;
  int qb = blk & 7;
  int h  = (blk >> 3) % HH;
  int b  = blk / (8 * HH);
  int q0 = qb * 128 + wave * 16;

  const size_t ld = 3 * DD;

  // q A-fragments (dh steps 0..31, 32..63) straight from global
  v16bf qa0, qa1;
  {
    const __bf16* qp = qkv + (size_t)(b * NNN + q0 + l16) * ld + h * DHH;
    int d0 = half * 16;
    reinterpret_cast<uint4*>(&qa0)[0] = *reinterpret_cast<const uint4*>(qp + d0);
    reinterpret_cast<uint4*>(&qa0)[1] = *reinterpret_cast<const uint4*>(qp + d0 + 8);
    reinterpret_cast<uint4*>(&qa1)[0] = *reinterpret_cast<const uint4*>(qp + 32 + d0);
    reinterpret_cast<uint4*>(&qa1)[1] = *reinterpret_cast<const uint4*>(qp + 32 + d0 + 8);
  }

  v8f o0 = {}, o1 = {}, o2 = {}, o3 = {};
  float mrow[8], lrow[8];
#pragma unroll
  for (int r = 0; r < 8; ++r) { mrow[r] = -3.0e38f; lrow[r] = 0.f; }

  const float scale2 = 0.125f * 1.44269504088896341f;  // (1/sqrt(64)) * log2(e)
  const float log2e  = 1.44269504088896341f;
  __bf16* sPw = sP + wave * (16 * 32);

  int key_row = tid >> 3;        // 0..31
  int dh0     = (tid & 7) << 3;  // 0..56

  for (int kt0 = 0; kt0 < NNN; kt0 += 32) {
    // cooperative K/V tile stage
    {
      const __bf16* kp = qkv + (size_t)(b * NNN + kt0 + key_row) * ld + DD + h * DHH + dh0;
      *reinterpret_cast<uint4*>(&sK[key_row * 64 + dh0]) = *reinterpret_cast<const uint4*>(kp);
      const __bf16* vp = kp + DD;
      union { uint4 q; __bf16 e[8]; } vu;
      vu.q = *reinterpret_cast<const uint4*>(vp);
      if (kt0 + 32 < NNN) {
        __builtin_prefetch(kp + 32 * ld, 0, 1);
        __builtin_prefetch(vp + 32 * ld, 0, 1);
      }
#pragma unroll
      for (int j = 0; j < 8; ++j) sVt[(dh0 + j) * 32 + key_row] = vu.e[j];
    }
    __syncthreads();

    // scores = q @ k^T  (two key n-tiles, two dh k-steps)
    v8f s0 = {}, s1 = {};
    {
      v16bf kb;
      kb = *reinterpret_cast<const v16bf*>(&sK[l16 * 64 + half * 16]);
      s0 = __builtin_amdgcn_wmma_f32_16x16x32_bf16(false, qa0, false, kb, (short)0, s0, false, false);
      kb = *reinterpret_cast<const v16bf*>(&sK[l16 * 64 + 32 + half * 16]);
      s0 = __builtin_amdgcn_wmma_f32_16x16x32_bf16(false, qa1, false, kb, (short)0, s0, false, false);
      kb = *reinterpret_cast<const v16bf*>(&sK[(16 + l16) * 64 + half * 16]);
      s1 = __builtin_amdgcn_wmma_f32_16x16x32_bf16(false, qa0, false, kb, (short)0, s1, false, false);
      kb = *reinterpret_cast<const v16bf*>(&sK[(16 + l16) * 64 + 32 + half * 16]);
      s1 = __builtin_amdgcn_wmma_f32_16x16x32_bf16(false, qa1, false, kb, (short)0, s1, false, false);
    }

    // scale + adjacency bias (base-2 domain), online softmax; row lives in 16 lanes
#pragma unroll
    for (int r = 0; r < 8; ++r) {
      int qr = q0 + r + 8 * half;
      float a0 = adj[(size_t)qr * NNN + kt0 + l16];
      float a1 = adj[(size_t)qr * NNN + kt0 + 16 + l16];
      float t0 = s0[r] * scale2 + a0 * log2e;
      float t1 = s1[r] * scale2 + a1 * log2e;
      float mx = row16_max(fmaxf(t0, t1));
      float mn = fmaxf(mrow[r], mx);
      float alpha = __builtin_amdgcn_exp2f(mrow[r] - mn);
      float p0 = __builtin_amdgcn_exp2f(t0 - mn);
      float p1 = __builtin_amdgcn_exp2f(t1 - mn);
      float rs = row16_sum(p0 + p1);
      lrow[r] = lrow[r] * alpha + rs;
      mrow[r] = mn;
      o0[r] *= alpha; o1[r] *= alpha; o2[r] *= alpha; o3[r] *= alpha;
      int prow = r + 8 * half;
      sPw[prow * 32 + l16]      = (__bf16)p0;
      sPw[prow * 32 + 16 + l16] = (__bf16)p1;
    }

    // attn += P @ V  (same-wave LDS round-trip re-lays P into A-fragment)
    v16bf pa;
    {
      const __bf16* pp = &sPw[l16 * 32 + half * 16];
      reinterpret_cast<uint4*>(&pa)[0] = *reinterpret_cast<const uint4*>(pp);
      reinterpret_cast<uint4*>(&pa)[1] = *reinterpret_cast<const uint4*>(pp + 8);
    }
    v16bf vb;
    vb = *reinterpret_cast<const v16bf*>(&sVt[(0 * 16 + l16) * 32 + half * 16]);
    o0 = __builtin_amdgcn_wmma_f32_16x16x32_bf16(false, pa, false, vb, (short)0, o0, false, false);
    vb = *reinterpret_cast<const v16bf*>(&sVt[(1 * 16 + l16) * 32 + half * 16]);
    o1 = __builtin_amdgcn_wmma_f32_16x16x32_bf16(false, pa, false, vb, (short)0, o1, false, false);
    vb = *reinterpret_cast<const v16bf*>(&sVt[(2 * 16 + l16) * 32 + half * 16]);
    o2 = __builtin_amdgcn_wmma_f32_16x16x32_bf16(false, pa, false, vb, (short)0, o2, false, false);
    vb = *reinterpret_cast<const v16bf*>(&sVt[(3 * 16 + l16) * 32 + half * 16]);
    o3 = __builtin_amdgcn_wmma_f32_16x16x32_bf16(false, pa, false, vb, (short)0, o3, false, false);

    __syncthreads();
  }

  // normalize and emit [B,N,H,DH] -> [token][D] bf16
#pragma unroll
  for (int r = 0; r < 8; ++r) {
    float inv = 1.0f / lrow[r];
    int tok = b * NNN + q0 + r + 8 * half;
    __bf16* op = attn_out + (size_t)tok * DD + h * DHH;
    op[0 * 16 + l16] = (__bf16)(o0[r] * inv);
    op[1 * 16 + l16] = (__bf16)(o1[r] * inv);
    op[2 * 16 + l16] = (__bf16)(o2[r] * inv);
    op[3 * 16 + l16] = (__bf16)(o3[r] * inv);
  }
}

extern "C" void kernel_launch(void* const* d_in, const int* in_sizes, int n_in,
                              void* d_out, int out_size, void* d_ws, size_t ws_size,
                              hipStream_t stream) {
  (void)in_sizes; (void)n_in; (void)out_size; (void)ws_size;
  const float* x      = (const float*)d_in[0];
  const float* adj    = (const float*)d_in[1];
  const float* ln1_g  = (const float*)d_in[2];
  const float* ln1_b  = (const float*)d_in[3];
  const float* qkv_w  = (const float*)d_in[4];
  const float* qkv_b  = (const float*)d_in[5];
  const float* proj_w = (const float*)d_in[6];
  const float* proj_b = (const float*)d_in[7];
  const float* ln2_g  = (const float*)d_in[8];
  const float* ln2_b  = (const float*)d_in[9];
  const float* fc1_w  = (const float*)d_in[10];
  const float* fc1_b  = (const float*)d_in[11];
  const float* fc2_w  = (const float*)d_in[12];
  const float* fc2_b  = (const float*)d_in[13];
  float* out = (float*)d_out;

  char* ws = (char*)d_ws;
  size_t off = 0;
  auto alloc = [&](size_t bytes) {
    void* p = ws + off;
    off += (bytes + 255) & ~(size_t)255;
    return p;
  };
  __bf16* qkvw_bf  = (__bf16*)alloc((size_t)DD * 3 * DD * 2);
  __bf16* projw_bf = (__bf16*)alloc((size_t)DD * DD * 2);
  __bf16* fc1w_bf  = (__bf16*)alloc((size_t)DD * HID * 2);
  __bf16* fc2w_bf  = (__bf16*)alloc((size_t)HID * DD * 2);
  __bf16* h_bf     = (__bf16*)alloc((size_t)TT * DD * 2);
  __bf16* qkv_bf   = (__bf16*)alloc((size_t)TT * 3 * DD * 2);
  __bf16* attn_bf  = (__bf16*)alloc((size_t)TT * DD * 2);
  float*  x1       = (float*) alloc((size_t)TT * DD * 4);
  __bf16* h2_bf    = (__bf16*)alloc((size_t)TT * DD * 2);
  __bf16* m1_bf    = (__bf16*)alloc((size_t)TT * HID * 2);

  int n;
  n = DD * 3 * DD; cvt_bf16_kernel<<<(n + 255) / 256, 256, 0, stream>>>(qkv_w,  qkvw_bf,  n);
  n = DD * DD;     cvt_bf16_kernel<<<(n + 255) / 256, 256, 0, stream>>>(proj_w, projw_bf, n);
  n = DD * HID;    cvt_bf16_kernel<<<(n + 255) / 256, 256, 0, stream>>>(fc1_w,  fc1w_bf,  n);
  n = HID * DD;    cvt_bf16_kernel<<<(n + 255) / 256, 256, 0, stream>>>(fc2_w,  fc2w_bf,  n);

  ln_kernel<<<TT, 256, 0, stream>>>(x, ln1_g, ln1_b, h_bf);

  gemm_wmma_kernel<0><<<dim3(3 * DD / 64, TT / 128), 256, 0, stream>>>(
      h_bf, qkvw_bf, qkv_b, nullptr, qkv_bf, TT, 3 * DD, DD);

  attn_kernel<<<BB * HH * (NNN / 128), 256, 0, stream>>>(qkv_bf, adj, attn_bf);

  gemm_wmma_kernel<2><<<dim3(DD / 64, TT / 128), 256, 0, stream>>>(
      attn_bf, projw_bf, proj_b, x, x1, TT, DD, DD);

  ln_kernel<<<TT, 256, 0, stream>>>(x1, ln2_g, ln2_b, h2_bf);

  gemm_wmma_kernel<1><<<dim3(HID / 64, TT / 128), 256, 0, stream>>>(
      h2_bf, fc1w_bf, fc1_b, nullptr, m1_bf, TT, HID, DD);

  gemm_wmma_kernel<2><<<dim3(DD / 64, TT / 128), 256, 0, stream>>>(
      m1_bf, fc2w_bf, fc2_b, x1, out, TT, DD, HID);
}
